// NTXentLoss_88613765251096
// MI455X (gfx1250) — matrix operand order repaired and checked by reading, not compiled, not run on hardware
//
#include <hip/hip_runtime.h>
#include <math.h>

typedef __attribute__((ext_vector_type(16))) _Float16 v16h;
typedef __attribute__((ext_vector_type(8)))  _Float16 v8h;
typedef __attribute__((ext_vector_type(4)))  _Float16 v4h;
typedef __attribute__((ext_vector_type(8)))  float    v8f;
typedef __attribute__((ext_vector_type(4)))  float    v4f;

#define NHALF    4096
#define NROWS    8192            // 2N
#define NTILES   (NROWS / 16)    // 512 column tiles
#define DDIM     256
#define TEMP_INV 10.0f           // 1 / 0.1
#define L2E      1.44269504088896340736f
#define LN2      0.69314718055994530942f
#define FIXED_M  (TEMP_INV * L2E)   // |cos|/T <= 10 -> base-2 logit bound

#define NBUF     4                   // LDS panel buffers (pipeline depth)
#define ROWH     264                 // 256 halfs + 8 pad halfs (bank-conflict-free)
#define PANEL_HALFS (16 * ROWH)      // one 16-column panel in LDS

#define SCHED_MFMA    0x008          // MFMA/WMMA group mask
#define SCHED_DS_READ 0x100          // DS-read group mask

// ---------------------------------------------------------------------------
// Kernel 1: row-normalize z = [z_i; z_j] and emit f16 zn (one wave per row)
// ---------------------------------------------------------------------------
__global__ __launch_bounds__(256) void nt_normalize(const float* __restrict__ zi,
                                                    const float* __restrict__ zj,
                                                    _Float16* __restrict__ zn) {
    const int row  = (blockIdx.x * blockDim.x + threadIdx.x) >> 5;   // 8 rows/block
    const int lane = threadIdx.x & 31;

    const float* src = (row < NHALF) ? (zi + (size_t)row * DDIM)
                                     : (zj + (size_t)(row - NHALF) * DDIM);
    v4f a = *(const v4f*)(src + lane * 4);
    v4f b = *(const v4f*)(src + 128 + lane * 4);

    float ss = 0.0f;
    #pragma unroll
    for (int i = 0; i < 4; ++i) ss += a[i] * a[i] + b[i] * b[i];
    #pragma unroll
    for (int off = 16; off > 0; off >>= 1) ss += __shfl_xor(ss, off, 32);

    const float inv = 1.0f / fmaxf(sqrtf(ss), 1e-8f);

    v4h ha, hb;
    #pragma unroll
    for (int i = 0; i < 4; ++i) {
        ha[i] = (_Float16)(a[i] * inv);
        hb[i] = (_Float16)(b[i] * inv);
    }
    _Float16* dst = zn + (size_t)row * DDIM;
    *(v4h*)(dst + lane * 4)       = ha;
    *(v4h*)(dst + 128 + lane * 4) = hb;
}

// ---------------------------------------------------------------------------
// async prefetch of one 16-col x 256-K f16 panel into a padded LDS slot.
// 512 B/row, 16 rows -> 512 chunks of 16 B; 256 threads x 2 chunks.
// ---------------------------------------------------------------------------
__device__ __forceinline__ void prefetch_panel(const _Float16* __restrict__ zn,
                                               _Float16* sSlot, int jt, int t) {
    #pragma unroll
    for (int i = 0; i < 2; ++i) {
        const int c   = t + i * 256;          // chunk id 0..511
        const int row = c >> 5;               // 0..15
        const int off = (c & 31) * 16;        // byte offset within 512B row
        const void* g = (const char*)(zn + (size_t)(jt * 16 + row) * DDIM) + off;
        const unsigned int l =
            (unsigned int)(size_t)((char*)sSlot + row * (ROWH * 2) + off);
        asm volatile("global_load_async_to_lds_b128 %0, %1, off"
                     :: "v"(l), "v"(g) : "memory");
    }
}

// ---------------------------------------------------------------------------
// Kernel 2: fused Gram + flash log-softmax.
// LDS-staged B panels (async engine, depth-4 pipeline), A register-resident,
// 8x v_wmma_f32_16x16x32_f16 per 16x16 tile, bounded-softmax accumulation.
// ---------------------------------------------------------------------------
__global__ __launch_bounds__(256) void nt_main(const _Float16* __restrict__ zn,
                                               float* __restrict__ row_nll) {
    __shared__ _Float16 sB[NBUF * PANEL_HALFS];

    const int t    = threadIdx.x;
    const int lane = t & 31;
    const int wave = t >> 5;
    const int tile = blockIdx.x * 8 + wave;   // 0..511 (row tile)
    const int rb   = tile * 16;
    const int lrow = lane & 15;
    const int hi   = lane >> 4;

    // ---- resident A fragments (ISA 16-bit A 16x32 layout) ----
    const _Float16* arow = zn + (size_t)(rb + lrow) * DDIM;
    v16h A[8];
    #pragma unroll
    for (int kc = 0; kc < 8; ++kc) {
        const int kb = kc * 32 + hi * 8;
        v8h lo  = *(const v8h*)(arow + kb);
        v8h hiv = *(const v8h*)(arow + kb + 16);
        v16h af;
        #pragma unroll
        for (int i = 0; i < 8; ++i) { af[i] = lo[i]; af[8 + i] = hiv[i]; }
        A[kc] = af;
    }

    // ---- sentinel column tiles: lane owns row v iff lrow == hi*8+v ----
    int dT[8], pT[8];
    #pragma unroll
    for (int v = 0; v < 8; ++v) {
        const bool own = (lrow == hi * 8 + v);
        dT[v] = own ? tile : -1;                    // diagonal tile
        pT[v] = own ? (tile ^ (NTILES / 2)) : -1;   // positive: (tile+256)%512
    }

    float s[8], p2[8];
    #pragma unroll
    for (int v = 0; v < 8; ++v) { s[v] = 0.0f; p2[v] = -1.0e30f; }

    const float scale2 = TEMP_INV * L2E;

    // ---- prologue: fill pipeline with panels 0..NBUF-2 ----
    #pragma unroll
    for (int p = 0; p < NBUF - 1; ++p)
        prefetch_panel(zn, sB + p * PANEL_HALFS, p, t);

    #pragma unroll 1
    for (int jt = 0; jt < NTILES; ++jt) {
        // my copies for panel jt complete (in-order: (NBUF-2)*2 = 4 may remain)
        asm volatile("s_wait_asynccnt 4" ::: "memory");
        __syncthreads();   // all waves' panel-jt copies done; slot (jt-1)%NBUF free

        // prefetch panel jt+NBUF-1 into the slot freed last iteration
        // (wrap-masked tail keeps the wait count uniform; extra panels land in
        //  slots that are never read again)
        const int pf = (jt + NBUF - 1) & (NTILES - 1);
        prefetch_panel(zn, sB + ((jt + NBUF - 1) & (NBUF - 1)) * PANEL_HALFS, pf, t);

        // ---- consume slot jt%NBUF ----
        const _Float16* bp = sB + (jt & (NBUF - 1)) * PANEL_HALFS
                                + lrow * ROWH + hi * 8;
        v8f c = {};
        #pragma unroll
        for (int kc = 0; kc < 8; ++kc) {
            v8h lo  = *(const v8h*)(bp + kc * 32);
            v8h hiv = *(const v8h*)(bp + kc * 32 + 16);
            v16h bf;
            #pragma unroll
            for (int i = 0; i < 8; ++i) { bf[i] = lo[i]; bf[8 + i] = hiv[i]; }
            c = __builtin_amdgcn_wmma_f32_16x16x32_f16(
                    false, A[kc], false, bf, (short)0, c, false, false);
        }
        // force schedule: issue all 16 DS reads, then the 8 WMMAs ->
        // distinct dest registers, staggered partial s_wait_dscnt instead of
        // a full-stop wait before every WMMA.
        __builtin_amdgcn_sched_group_barrier(SCHED_DS_READ, 16, 0);
        __builtin_amdgcn_sched_group_barrier(SCHED_MFMA,     8, 0);

        #pragma unroll
        for (int v = 0; v < 8; ++v) {
            const float arg = __builtin_fmaf(c[v], scale2, -FIXED_M);
            const float e   = (jt == dT[v]) ? 0.0f : __builtin_amdgcn_exp2f(arg);
            s[v] += e;
            p2[v] = (jt == pT[v]) ? arg : p2[v];
        }
    }

    // ---- reduce s (sum) and p2 (max) across each 16-lane half ----
    #pragma unroll
    for (int off = 1; off < 16; off <<= 1) {
        #pragma unroll
        for (int v = 0; v < 8; ++v) {
            s[v]  += __shfl_xor(s[v], off, 32);
            p2[v]  = fmaxf(p2[v], __shfl_xor(p2[v], off, 32));
        }
    }

    if (lrow == 0) {
        #pragma unroll
        for (int v = 0; v < 8; ++v) {
            // nll = (M + log2(s)) - (p2 + M) in base-2, times ln2
            row_nll[rb + hi * 8 + v] =
                (__builtin_amdgcn_logf(s[v]) - p2[v]) * LN2;
        }
    }
}

// ---------------------------------------------------------------------------
// Kernel 3: mean over 8192 per-row NLLs -> scalar
// ---------------------------------------------------------------------------
__global__ __launch_bounds__(256) void nt_reduce(const float* __restrict__ row_nll,
                                                 float* __restrict__ out) {
    float acc = 0.0f;
    for (int i = threadIdx.x; i < NROWS; i += 256) acc += row_nll[i];

    #pragma unroll
    for (int off = 16; off > 0; off >>= 1) acc += __shfl_xor(acc, off, 32);

    __shared__ float sm[8];
    const int wave = threadIdx.x >> 5;
    const int lane = threadIdx.x & 31;
    if (lane == 0) sm[wave] = acc;
    __syncthreads();
    if (wave == 0) {
        float tv = (lane < 8) ? sm[lane] : 0.0f;
        #pragma unroll
        for (int off = 4; off > 0; off >>= 1) tv += __shfl_xor(tv, off, 32);
        if (lane == 0) out[0] = tv * (1.0f / (float)NROWS);
    }
}

// ---------------------------------------------------------------------------
extern "C" void kernel_launch(void* const* d_in, const int* in_sizes, int n_in,
                              void* d_out, int out_size, void* d_ws, size_t ws_size,
                              hipStream_t stream) {
    const float* zi = (const float*)d_in[0];
    const float* zj = (const float*)d_in[1];

    _Float16* zn      = (_Float16*)d_ws;                                   // 4 MB
    float*    row_nll = (float*)((char*)d_ws +
                                 (size_t)NROWS * DDIM * sizeof(_Float16)); // +32 KB

    nt_normalize<<<NROWS / 8, 256, 0, stream>>>(zi, zj, zn);
    nt_main<<<NTILES / 8, 256, 0, stream>>>(zn, row_nll);
    nt_reduce<<<1, 256, 0, stream>>>(row_nll, (float*)d_out);
}
